// DefConvLayer_red_18605798326571
// MI455X (gfx1250) — matrix operand order repaired and checked by reading, not compiled, not run on hardware
//
#include <hip/hip_runtime.h>
#include <hip/hip_bf16.h>

// ---- types for WMMA operands -------------------------------------------------
typedef _Float16 v16h __attribute__((ext_vector_type(16)));
typedef _Float16 v8h  __attribute__((ext_vector_type(8)));
typedef _Float16 v4h  __attribute__((ext_vector_type(4)));
typedef float    v8f  __attribute__((ext_vector_type(8)));

// Problem constants (fixed by the reference).
#define BATCH   8
#define HW      128          // Hi = Wi = Ho = Wo
#define CIN     64
#define NTAPS   9
#define FILT    128
#define KDIM    (NTAPS * CIN)        // 576
#define ASTRIDE 584                  // padded K stride in f16: 16B-aligned rows,
                                     // 292 dwords -> 16 lanes hit disjoint bank windows
#define PIX_PER_WG 128               // one (m,h) row of output pixels per workgroup

// ---- LDS fragment loaders ----------------------------------------------------
// A-matrix 16x32 f16 layout (ISA 7.12.2):
//   lanes 0-15  : row M=lane,    K = {0..7, 16..23}
//   lanes 16-31 : row M=lane-16, K = {8..15, 24..31}
__device__ __forceinline__ v16h load_a_frag(const _Float16* __restrict__ base,
                                            int lane, int k0) {
  const int row  = lane & 15;
  const int ksel = (lane >> 4) << 3;                 // 0 or 8
  const _Float16* p = base + row * ASTRIDE + k0 + ksel;
  v8h lo = *(const v8h*)(p);                         // K +0..+7
  v8h hi = *(const v8h*)(p + 16);                    // K +16..+23
  v16h r;
#pragma unroll
  for (int i = 0; i < 8; ++i) { r[i] = lo[i]; r[i + 8] = hi[i]; }
  return r;
}

// B-matrix 32x16 f16 layout (same K order as A):
//   lanes 0-15  : column N=lane,    K = {0..7, 16..23}
//   lanes 16-31 : column N=lane-16, K = {8..15, 24..31}
// With Wlds stored column-major [f][k], each lane reads two 8-half runs.
__device__ __forceinline__ v16h load_b_frag(const _Float16* __restrict__ base,
                                            int lane, int k0) {
  const int col  = lane & 15;
  const int koff = (lane >> 4) << 3;                 // 0 or 8
  const _Float16* p = base + col * ASTRIDE + k0 + koff;
  v8h lo = *(const v8h*)(p);
  v8h hi = *(const v8h*)(p + 16);
  v16h r;
#pragma unroll
  for (int i = 0; i < 8; ++i) { r[i] = lo[i]; r[i + 8] = hi[i]; }
  return r;
}

// ---- kernel ------------------------------------------------------------------
__global__ __launch_bounds__(256)
void defconv_wmma_kernel(const float* __restrict__ x,
                         const float* __restrict__ offs,
                         const float* __restrict__ W,
                         float* __restrict__ out) {
  extern __shared__ _Float16 smem[];
  _Float16* Alds = smem;                        // [128 px][ASTRIDE]  sampled f16
  _Float16* Wlds = smem + PIX_PER_WG * ASTRIDE; // [128 f ][ASTRIDE]  W^T f16

  const int tid = threadIdx.x;
  const int mh  = blockIdx.x;                   // m*128 + h
  const int h   = mh & (HW - 1);

  // ---- Phase 0: pack W (c,n,f) fp32 -> Wlds[f][k=n*64+c] f16 ----------------
  for (int idx = tid; idx < FILT * KDIM; idx += 256) {
    const int f = idx & (FILT - 1);
    const int k = idx >> 7;
    const int c = k & (CIN - 1);
    const int n = k >> 6;
    Wlds[f * ASTRIDE + k] = (_Float16)W[(c * NTAPS + n) * FILT + f];
  }

  // ---- Phase 1: bilinear sampling -> Alds[px][k] f16 ------------------------
  {
    const int p     = tid >> 1;                 // pixel w in 0..127
    const int cbase = (tid & 1) << 5;           // channel half: 0 or 32
    const float* xm   = x + (size_t)(mh >> 7) * (HW * HW * CIN);
    const float* offp = offs + ((size_t)mh * HW + p) * (2 * NTAPS);
    const float hh = (float)h, ww = (float)p;
    _Float16* arow = Alds + p * ASTRIDE + cbase;

    for (int n = 0; n < NTAPS; ++n) {
      const float ci = fminf(fmaxf(hh + offp[2 * n + 0], 0.f), (float)(HW - 1));
      const float cj = fminf(fmaxf(ww + offp[2 * n + 1], 0.f), (float)(HW - 1));
      const float fi0 = floorf(ci), fj0 = floorf(cj);
      const int i0 = (int)fi0, j0 = (int)fj0;
      int i1 = i0 + 1; if (i1 > HW - 1) i1 = HW - 1;   // == ceil given frac==0 there
      int j1 = j0 + 1; if (j1 > HW - 1) j1 = HW - 1;
      const float fi = ci - fi0, fj = cj - fj0;

      const float* p00 = xm + (((i0 << 7) + j0) << 6) + cbase;
      const float* p01 = xm + (((i0 << 7) + j1) << 6) + cbase;
      const float* p10 = xm + (((i1 << 7) + j0) << 6) + cbase;
      const float* p11 = xm + (((i1 << 7) + j1) << 6) + cbase;
      _Float16* dst = arow + (n << 6);

#pragma unroll
      for (int cc = 0; cc < 32; cc += 4) {
        const float4 v00 = *(const float4*)(p00 + cc);
        const float4 v01 = *(const float4*)(p01 + cc);
        const float4 v10 = *(const float4*)(p10 + cc);
        const float4 v11 = *(const float4*)(p11 + cc);
        v4h o;
        {
          const float t = fmaf(v01.x - v00.x, fj, v00.x);
          const float b = fmaf(v11.x - v10.x, fj, v10.x);
          o[0] = (_Float16)fmaf(b - t, fi, t);
        }
        {
          const float t = fmaf(v01.y - v00.y, fj, v00.y);
          const float b = fmaf(v11.y - v10.y, fj, v10.y);
          o[1] = (_Float16)fmaf(b - t, fi, t);
        }
        {
          const float t = fmaf(v01.z - v00.z, fj, v00.z);
          const float b = fmaf(v11.z - v10.z, fj, v10.z);
          o[2] = (_Float16)fmaf(b - t, fi, t);
        }
        {
          const float t = fmaf(v01.w - v00.w, fj, v00.w);
          const float b = fmaf(v11.w - v10.w, fj, v10.w);
          o[3] = (_Float16)fmaf(b - t, fi, t);
        }
        *(v4h*)(dst + cc) = o;                 // ds_store_b64, 8B aligned
      }
    }
  }

  __syncthreads();

  // ---- Phase 2: WMMA GEMM  out[128px,128f] = Alds(128x576) * Wlds^T ---------
  const int lane = tid & 31;
  const int wv   = tid >> 5;                   // 8 waves
  const int wm   = wv >> 1;                    // 0..3 : 32-pixel slab
  const int wn   = wv & 1;                     // 0..1 : 64-filter slab

  v8f acc[2][4] = {};                          // 2 M-tiles x 4 N-tiles of 16x16

  const _Float16* Abase0 = Alds + (wm * 32) * ASTRIDE;
  const _Float16* Abase1 = Abase0 + 16 * ASTRIDE;

  for (int k0 = 0; k0 < KDIM; k0 += 32) {
    const v16h a0 = load_a_frag(Abase0, lane, k0);
    const v16h a1 = load_a_frag(Abase1, lane, k0);
#pragma unroll
    for (int t = 0; t < 4; ++t) {
      const v16h bm = load_b_frag(Wlds + (wn * 64 + t * 16) * ASTRIDE, lane, k0);
      acc[0][t] = __builtin_amdgcn_wmma_f32_16x16x32_f16(
          false, a0, false, bm, (short)0, acc[0][t], false, false);
      acc[1][t] = __builtin_amdgcn_wmma_f32_16x16x32_f16(
          false, a1, false, bm, (short)0, acc[1][t], false, false);
    }
  }

  // ---- store: 16x16 f32 D layout (VGPR v -> M row, lane -> N col) -----------
  float* outp = out + ((size_t)mh << 14);      // pixel base * 128 filters
  const int nrow = (lane >> 4) << 3;           // lanes 16-31 hold M+8
  const int ncol = lane & 15;
#pragma unroll
  for (int mt = 0; mt < 2; ++mt) {
#pragma unroll
    for (int t = 0; t < 4; ++t) {
      const int R = wm * 32 + mt * 16 + nrow;
      const int F = wn * 64 + t * 16 + ncol;
      float* o = outp + (size_t)R * FILT + F;
#pragma unroll
      for (int v = 0; v < 8; ++v)
        o[(size_t)v * FILT] = acc[mt][t][v];
    }
  }
}

// ---- host-side launch --------------------------------------------------------
extern "C" void kernel_launch(void* const* d_in, const int* in_sizes, int n_in,
                              void* d_out, int out_size, void* d_ws, size_t ws_size,
                              hipStream_t stream) {
  (void)in_sizes; (void)n_in; (void)out_size; (void)d_ws; (void)ws_size;
  const float* x    = (const float*)d_in[0];
  const float* offs = (const float*)d_in[1];
  const float* W    = (const float*)d_in[2];
  float* out        = (float*)d_out;

  const size_t smem_bytes = (size_t)2 * PIX_PER_WG * ASTRIDE * sizeof(_Float16); // 292KB
  // Allow >default dynamic LDS (CDNA5 WGP supports up to 320KB per workgroup).
  (void)hipFuncSetAttribute((const void*)defconv_wmma_kernel,
                            hipFuncAttributeMaxDynamicSharedMemorySize,
                            (int)smem_bytes);

  const int num_wgs = BATCH * HW;              // one (m,h) row per workgroup = 1024
  defconv_wmma_kernel<<<dim3(num_wgs), dim3(256), smem_bytes, stream>>>(x, offs, W, out);
}
